// EMMatchHeadPlus_36919538876995
// MI455X (gfx1250) — matrix operand order repaired and checked by reading, not compiled, not run on hardware
//
#include <hip/hip_runtime.h>
#include <hip/hip_bf16.h>
#include <stdint.h>

// MI455X (gfx1250): implicit-GEMM convs on v_wmma_f32_16x16x32_f16 with
// double-buffered LDS tiles and async global->LDS weight staging.
// EM/softmax stages on VALU. wave32, 256-thread blocks.

#define R_    128
#define C_    256
#define H_    28
#define W_    28
#define N_    784      // 28*28
#define CIN1  259      // C + 3 (pos_z, neg_z, m_a)
#define KSTR1 288      // cin padded to multiple of 32 for conv1 weight tiles
#define NN_   3136     // 56*56

typedef __attribute__((ext_vector_type(16))) _Float16 v16h;
typedef __attribute__((ext_vector_type(4)))  _Float16 v4h;
typedef __attribute__((ext_vector_type(8)))  float    v8f;
typedef __attribute__((ext_vector_type(4)))  int      v4i;

#if defined(__has_builtin)
  #if __has_builtin(__builtin_amdgcn_global_load_async_to_lds_b128)
    #define HAVE_ASYNC_LDS 1
  #else
    #define HAVE_ASYNC_LDS 0
  #endif
  #if __has_builtin(__builtin_amdgcn_s_wait_asynccnt)
    #define HAVE_WAIT_ASYNC 1
  #else
    #define HAVE_WAIT_ASYNC 0
  #endif
#else
  #define HAVE_ASYNC_LDS 0
  #define HAVE_WAIT_ASYNC 0
#endif

__device__ __forceinline__ void async_copy16(const _Float16* src, _Float16* dst) {
#if HAVE_ASYNC_LDS
  // signature: (v4i addrspace(1)* gsrc, v4i addrspace(3)* lds, imm offset, imm cpol)
  __builtin_amdgcn_global_load_async_to_lds_b128(
      (__attribute__((address_space(1))) v4i*)src,
      (__attribute__((address_space(3))) v4i*)dst, 0, 0);
#else
  const uint32_t* s32 = (const uint32_t*)src;
  uint32_t* d32 = (uint32_t*)dst;
  #pragma unroll
  for (int j = 0; j < 4; ++j) d32[j] = s32[j];
#endif
}

__device__ __forceinline__ void wait_async_stage() {
#if HAVE_ASYNC_LDS
  #if HAVE_WAIT_ASYNC
    __builtin_amdgcn_s_wait_asynccnt(0);
  #else
    asm volatile("s_wait_asynccnt 0" ::: "memory");
  #endif
#endif
}

// ---------------------------------------------------------------- utilities

__device__ __forceinline__ void rs_taps(int o, int* idx, float* wgt) {
  // antialiased bilinear 2x downsample taps (jax.image.resize default):
  // output o samples input at 2o+0.5, triangle radius 2 -> weights .125/.375/.375/.125
  const float base0 = 0.125f, base1 = 0.375f;
  float s = 0.f;
  #pragma unroll
  for (int t = 0; t < 4; ++t) {
    int j = 2 * o - 1 + t;
    float w = (t == 0 || t == 3) ? base0 : base1;
    if (j >= 0 && j < 56) { idx[t] = j; wgt[t] = w; s += w; }
    else                  { idx[t] = 0; wgt[t] = 0.f; }
  }
  float inv = 1.f / s;
  #pragma unroll
  for (int t = 0; t < 4; ++t) wgt[t] *= inv;
}

// ------------------------------------------------------------ small kernels

__global__ void resize_masks_kernel(const float* __restrict__ ma_in,
                                    const float* __restrict__ mb_in,
                                    float* __restrict__ ma_out,
                                    float* __restrict__ mb_out) {
  int idx = blockIdx.x * blockDim.x + threadIdx.x;
  if (idx >= 2 * R_ * N_) return;
  const float* src = ma_in; float* dst = ma_out;
  int i = idx;
  if (i >= R_ * N_) { src = mb_in; dst = mb_out; i -= R_ * N_; }
  int r = i / N_, p = i % N_;
  int oh = p / W_, ow = p % W_;
  int ih[4], iw[4]; float wh[4], ww[4];
  rs_taps(oh, ih, wh); rs_taps(ow, iw, ww);
  const float* s = src + (size_t)r * 56 * 56;
  float acc = 0.f;
  #pragma unroll
  for (int a = 0; a < 4; ++a)
    #pragma unroll
    for (int b = 0; b < 4; ++b)
      acc += wh[a] * ww[b] * s[ih[a] * 56 + iw[b]];
  dst[i] = acc;
}

__global__ void cvt_w1_kernel(const float* __restrict__ w, _Float16* __restrict__ wt) {
  int idx = blockIdx.x * blockDim.x + threadIdx.x;
  if (idx >= 256 * 9 * KSTR1) return;
  int co  = idx / (9 * KSTR1);
  int rem = idx % (9 * KSTR1);
  int t   = rem / KSTR1;
  int ci  = rem % KSTR1;
  float v = (ci < CIN1) ? w[((size_t)co * CIN1 + ci) * 9 + t] : 0.f;
  wt[idx] = (_Float16)v;
}

__global__ void cvt_ws_kernel(const float* __restrict__ w, _Float16* __restrict__ wt) {
  int idx = blockIdx.x * blockDim.x + threadIdx.x;
  if (idx >= 3 * 256 * 9 * 256) return;
  int l    = idx / (256 * 9 * 256);
  int rem  = idx % (256 * 9 * 256);
  int co   = rem / (9 * 256);
  int rem2 = rem % (9 * 256);
  int t    = rem2 / 256;
  int ci   = rem2 % 256;
  wt[idx] = (_Float16)w[(((size_t)l * 256 + co) * 256 + ci) * 9 + t];
}

__global__ void cvt_wd_kernel(const float* __restrict__ w, _Float16* __restrict__ wt) {
  int idx = blockIdx.x * blockDim.x + threadIdx.x;
  if (idx >= 4 * 256 * 256) return;
  int kl  = idx / (256 * 256);
  int rem = idx % (256 * 256);
  int o   = rem / 256;
  int i   = rem % 256;
  wt[idx] = (_Float16)w[((size_t)i * 256 + o) * 4 + kl];   // deconv_w[i][o][k][l]
}

__global__ void mu_init_kernel(const float* __restrict__ pos,
                               const float* __restrict__ neg,
                               float* __restrict__ mu) {
  int idx = blockIdx.x * blockDim.x + threadIdx.x;
  if (idx >= R_ * C_ * 8) return;
  int rem = idx % (C_ * 8);
  int c = rem / 8, k = rem % 8;
  mu[idx] = (k < 4) ? pos[c * 4 + k] : neg[c * 4 + (k - 4)];
}

__global__ void copy_featb_kernel(const float* __restrict__ fb, _Float16* __restrict__ feat) {
  int idx = blockIdx.x * blockDim.x + threadIdx.x;
  if (idx >= R_ * C_ * N_) return;
  int r = idx / (C_ * N_);
  int rem = idx % (C_ * N_);
  feat[(size_t)r * CIN1 * N_ + rem] = (_Float16)fb[idx];
}

__global__ __launch_bounds__(256)
void gap_kernel(const float* __restrict__ fa, const float* __restrict__ mA,
                float* __restrict__ gap) {
  int r = blockIdx.x, tid = threadIdx.x;
  __shared__ float sm[N_];
  __shared__ float red[256];
  float ps = 0.f;
  for (int p = tid; p < N_; p += 256) {
    float b = (mA[r * N_ + p] >= 0.f) ? 1.f : 0.f;
    sm[p] = b; ps += b;
  }
  red[tid] = ps; __syncthreads();
  for (int s = 128; s > 0; s >>= 1) { if (tid < s) red[tid] += red[tid + s]; __syncthreads(); }
  float inv = 1.f / (red[0] + 1e-5f);
  int c = tid;
  const float* row = fa + ((size_t)r * C_ + c) * N_;
  float s2 = 0.f;
  for (int p = 0; p < N_; ++p) s2 += row[p] * sm[p];
  gap[r * C_ + c] = s2 * inv;
}

__global__ __launch_bounds__(256)
void conv2gap_kernel(const float* __restrict__ gap, const float* __restrict__ w2,
                     const float* __restrict__ b2, const float* __restrict__ b1,
                     float* __restrict__ co2) {
  int r = blockIdx.x, co = threadIdx.x;
  __shared__ float g[256];
  g[co] = gap[r * C_ + co];
  __syncthreads();
  const float* wr = w2 + (size_t)co * C_;
  float s = b2[co] + b1[co];           // fold conv1_b + conv2_b into per-(r,co) bias
  for (int ci = 0; ci < C_; ++ci) s += wr[ci] * g[ci];
  co2[r * C_ + co] = s;
}

// -------------------------------------------------------------- EM stage

__global__ __launch_bounds__(256)
void em_stage_kernel(const float* __restrict__ x,      // [R,256,784]
                     const float* __restrict__ mres,   // [R,784] raw resized mask
                     float* __restrict__ mu) {         // [R,256,8]
  int r = blockIdx.x, pn = blockIdx.y, tid = threadIdx.x;
  int kb = pn * 4;
  __shared__ float sMu[C_][4];
  __shared__ float sZ[N_][4];
  __shared__ float sRed[256][4];
  for (int i = tid; i < C_; i += 256)
    #pragma unroll
    for (int k = 0; k < 4; ++k) sMu[i][k] = mu[((size_t)r * C_ + i) * 8 + kb + k];
  __syncthreads();
  // z = softmax_k(20 * x^T mu) * mask
  for (int n = tid; n < N_; n += 256) {
    float a0 = 0, a1 = 0, a2 = 0, a3 = 0;
    const float* xc = x + (size_t)r * C_ * N_ + n;
    for (int c = 0; c < C_; ++c) {
      float xv = xc[(size_t)c * N_];
      a0 += xv * sMu[c][0]; a1 += xv * sMu[c][1];
      a2 += xv * sMu[c][2]; a3 += xv * sMu[c][3];
    }
    float m0 = fmaxf(fmaxf(a0, a1), fmaxf(a2, a3));
    float e0 = __expf(20.f * (a0 - m0)), e1 = __expf(20.f * (a1 - m0));
    float e2 = __expf(20.f * (a2 - m0)), e3 = __expf(20.f * (a3 - m0));
    float inv = 1.f / (e0 + e1 + e2 + e3);
    float bin = (mres[r * N_ + n] >= 0.f) ? 1.f : 0.f;
    float f = inv * (pn ? (1.f - bin) : bin);
    sZ[n][0] = e0 * f; sZ[n][1] = e1 * f; sZ[n][2] = e2 * f; sZ[n][3] = e3 * f;
  }
  __syncthreads();
  // l1-normalize z over N
  float c0 = 0, c1 = 0, c2 = 0, c3 = 0;
  for (int n = tid; n < N_; n += 256) { c0 += sZ[n][0]; c1 += sZ[n][1]; c2 += sZ[n][2]; c3 += sZ[n][3]; }
  sRed[tid][0] = c0; sRed[tid][1] = c1; sRed[tid][2] = c2; sRed[tid][3] = c3;
  __syncthreads();
  for (int s = 128; s > 0; s >>= 1) {
    if (tid < s) { for (int k = 0; k < 4; ++k) sRed[tid][k] += sRed[tid + s][k]; }
    __syncthreads();
  }
  float i0 = 1.f / (1e-6f + sRed[0][0]), i1 = 1.f / (1e-6f + sRed[0][1]);
  float i2 = 1.f / (1e-6f + sRed[0][2]), i3 = 1.f / (1e-6f + sRed[0][3]);
  for (int n = tid; n < N_; n += 256) { sZ[n][0] *= i0; sZ[n][1] *= i1; sZ[n][2] *= i2; sZ[n][3] *= i3; }
  __syncthreads();
  // mu = l2norm_C(x z)
  {
    int c = tid;
    const float* xr = x + ((size_t)r * C_ + c) * N_;
    float a0 = 0, a1 = 0, a2 = 0, a3 = 0;
    for (int n = 0; n < N_; ++n) {
      float xv = xr[n];
      a0 += xv * sZ[n][0]; a1 += xv * sZ[n][1]; a2 += xv * sZ[n][2]; a3 += xv * sZ[n][3];
    }
    sMu[c][0] = a0; sMu[c][1] = a1; sMu[c][2] = a2; sMu[c][3] = a3;
    sRed[c][0] = a0 * a0; sRed[c][1] = a1 * a1; sRed[c][2] = a2 * a2; sRed[c][3] = a3 * a3;
  }
  __syncthreads();
  for (int s = 128; s > 0; s >>= 1) {
    if (tid < s) { for (int k = 0; k < 4; ++k) sRed[tid][k] += sRed[tid + s][k]; }
    __syncthreads();
  }
  float n0 = 1.f / (1e-6f + sqrtf(sRed[0][0])), n1 = 1.f / (1e-6f + sqrtf(sRed[0][1]));
  float n2 = 1.f / (1e-6f + sqrtf(sRed[0][2])), n3 = 1.f / (1e-6f + sqrtf(sRed[0][3]));
  {
    int c = tid;
    size_t b = ((size_t)r * C_ + c) * 8 + kb;
    mu[b + 0] = sMu[c][0] * n0; mu[b + 1] = sMu[c][1] * n1;
    mu[b + 2] = sMu[c][2] * n2; mu[b + 3] = sMu[c][3] * n3;
  }
}

// ------------------------------------------------- prop + feature assembly

__global__ __launch_bounds__(256)
void prop_feat_kernel(const float* __restrict__ fb, const float* __restrict__ mu,
                      const float* __restrict__ mA, _Float16* __restrict__ feat) {
  int r = blockIdx.x, tid = threadIdx.x;
  __shared__ float sMu[C_][8];
  for (int i = tid; i < C_; i += 256)
    #pragma unroll
    for (int k = 0; k < 8; ++k) sMu[i][k] = mu[((size_t)r * C_ + i) * 8 + k];
  __syncthreads();
  for (int p = tid; p < N_; p += 256) {
    float a[8] = {0, 0, 0, 0, 0, 0, 0, 0};
    const float* xc = fb + (size_t)r * C_ * N_ + p;
    for (int c = 0; c < C_; ++c) {
      float xv = xc[(size_t)c * N_];
      #pragma unroll
      for (int k = 0; k < 8; ++k) a[k] += xv * sMu[c][k];
    }
    float mx = a[0];
    #pragma unroll
    for (int k = 1; k < 8; ++k) mx = fmaxf(mx, a[k]);
    float z[8], s = 0.f;
    #pragma unroll
    for (int k = 0; k < 8; ++k) { z[k] = __expf(a[k] - mx); s += z[k]; }
    float inv = 1.f / s;
    float pz = (z[0] + z[1] + z[2] + z[3]) * inv;
    float nz = (z[4] + z[5] + z[6] + z[7]) * inv;
    size_t base = (size_t)r * CIN1 * N_;
    feat[base + (size_t)256 * N_ + p] = (_Float16)pz;
    feat[base + (size_t)257 * N_ + p] = (_Float16)nz;
    feat[base + (size_t)258 * N_ + p] = (_Float16)mA[r * N_ + p];
  }
}

// ---------------------------------------------- WMMA implicit-GEMM 3x3 conv
// Block owns all 16 co-tiles for one 16-position tile; each wave does 2 tiles
// (B fragment reused for 2 WMMAs). Double-buffered LDS: one barrier per
// k-step, next A tiles staged via async global->LDS while WMMA runs.

__global__ __launch_bounds__(256)
void conv3x3_wmma(const _Float16* __restrict__ in,   // [R][cin][784]
                  const _Float16* __restrict__ wt,   // [256][9][kstr], k contiguous
                  int cin, int kchunks, int kstr,
                  const float* __restrict__ bias, int bias_perR, int relu,
                  _Float16* __restrict__ out) {      // [R][256][784]
  const int ptile = blockIdx.x;          // 49 position tiles
  const int r     = blockIdx.y;
  const int tid   = threadIdx.x;
  const int wave  = tid >> 5;
  const int lane  = tid & 31;
  const int posbase = ptile * 16;

  __shared__ __align__(16) _Float16 sB[2][16][40];       // [buf][n][k]
  __shared__ __align__(16) _Float16 sA[2][16][16][40];   // [buf][tile][m][k]

  v8f acc[2];
  acc[0] = (v8f){0.f, 0.f, 0.f, 0.f, 0.f, 0.f, 0.f, 0.f};
  acc[1] = acc[0];

  const int S = 9 * kchunks;

  auto stage = [&](int step, int buf) {
    int t   = step / kchunks;
    int kc  = step % kchunks;
    int ci0 = kc * 32;
    int kh = t / 3 - 1, kw = t % 3 - 1;
    // B tile (VALU path: needs halo zero-fill)
    for (int i = tid; i < 16 * 32; i += 256) {
      int n = i & 15, k = i >> 4;
      int p = posbase + n;
      int h = p / W_ + kh, w = p % W_ + kw;
      int ci = ci0 + k;
      _Float16 v = (_Float16)0.f;
      if ((unsigned)h < (unsigned)H_ && (unsigned)w < (unsigned)W_ && ci < cin)
        v = in[((size_t)r * cin + ci) * N_ + h * W_ + w];
      sB[buf][n][k] = v;
    }
    // A tiles: 16 tiles x 16 rows x 4 segments of 8 halfs (16B, async DMA)
    for (int s2 = tid; s2 < 1024; s2 += 256) {
      int tile = s2 >> 6;
      int m    = (s2 >> 2) & 15;
      int seg  = s2 & 3;
      const _Float16* src = wt + ((size_t)(tile * 16 + m) * 9 + t) * kstr + ci0 + seg * 8;
      async_copy16(src, &sA[buf][tile][m][seg * 8]);
    }
  };

  stage(0, 0);
  wait_async_stage();
  __syncthreads();

  const int ml = lane & 15;
  const int hi = lane >> 4;

  for (int step = 0; step < S; ++step) {
    int buf = step & 1;
    if (step + 1 < S) stage(step + 1, buf ^ 1);
    // gather fragments per ISA 16-bit A(16x32) / B(32x16) lane layouts
    v16h bfrag;
    {
      const _Float16* br = &sB[buf][ml][0];
      int o2 = hi * 8;
      v4h* bp = (v4h*)&bfrag;
      bp[0] = *(const v4h*)(br + o2);
      bp[1] = *(const v4h*)(br + o2 + 4);
      bp[2] = *(const v4h*)(br + o2 + 16);
      bp[3] = *(const v4h*)(br + o2 + 20);
    }
    #pragma unroll
    for (int pass = 0; pass < 2; ++pass) {
      int tile = wave + pass * 8;
      const _Float16* ar = &sA[buf][tile][ml][0];
      int o1 = hi * 8;
      v16h afrag;
      v4h* ap = (v4h*)&afrag;
      ap[0] = *(const v4h*)(ar + o1);
      ap[1] = *(const v4h*)(ar + o1 + 4);
      ap[2] = *(const v4h*)(ar + o1 + 16);
      ap[3] = *(const v4h*)(ar + o1 + 20);
      acc[pass] = __builtin_amdgcn_wmma_f32_16x16x32_f16(false, afrag, false, bfrag,
                                                         (short)0, acc[pass], false, false);
    }
    wait_async_stage();
    __syncthreads();
  }

  // epilogue: lane holds column n = lane&15, rows v + 8*(lane>=16)
  int n = lane & 15;
  int mrow0 = hi * 8;
  int p = posbase + n;
  #pragma unroll
  for (int pass = 0; pass < 2; ++pass) {
    int coBase = (wave + pass * 8) * 16;
    #pragma unroll
    for (int v = 0; v < 8; ++v) {
      int co = coBase + mrow0 + v;
      float b = bias_perR ? bias[(size_t)r * C_ + co] : bias[co];
      float val = acc[pass][v] + b;
      if (relu) val = val > 0.f ? val : 0.f;
      out[((size_t)r * C_ + co) * N_ + p] = (_Float16)val;
    }
  }
}

// ---------------- deconv (4 disjoint 1x1 GEMMs) fused with ReLU + logits 1x1

__global__ __launch_bounds__(256)
void deconv_logits_wmma(const _Float16* __restrict__ in,   // [R][256][784]
                        const _Float16* __restrict__ wt,   // [4][o=256][i=256]
                        const float* __restrict__ dbias,   // [256]
                        const float* __restrict__ lw,      // logits_w [256]
                        const float* __restrict__ lb,      // logits_b [1]
                        float* __restrict__ outp) {        // [R][56*56]
  const int ptile = blockIdx.x;
  const int r     = blockIdx.y;
  const int kl    = blockIdx.z;          // 0..3 -> (kk,ll)
  const int tid   = threadIdx.x;
  const int wave  = tid >> 5;
  const int lane  = tid & 31;
  const int posbase = ptile * 16;

  __shared__ __align__(16) _Float16 sB[2][16][40];
  __shared__ __align__(16) _Float16 sA[2][16][16][40];
  __shared__ float sRed[8][16];

  v8f acc[2];
  acc[0] = (v8f){0.f, 0.f, 0.f, 0.f, 0.f, 0.f, 0.f, 0.f};
  acc[1] = acc[0];

  auto stage = [&](int kc, int buf) {
    int ci0 = kc * 32;
    for (int i = tid; i < 16 * 32; i += 256) {
      int n = i & 15, k = i >> 4;
      sB[buf][n][k] = in[((size_t)r * C_ + ci0 + k) * N_ + posbase + n];
    }
    for (int s2 = tid; s2 < 1024; s2 += 256) {
      int tile = s2 >> 6;
      int m    = (s2 >> 2) & 15;
      int seg  = s2 & 3;
      const _Float16* src = wt + ((size_t)kl * 256 + tile * 16 + m) * 256 + ci0 + seg * 8;
      async_copy16(src, &sA[buf][tile][m][seg * 8]);
    }
  };

  stage(0, 0);
  wait_async_stage();
  __syncthreads();

  const int ml = lane & 15;
  const int hi = lane >> 4;

  for (int kc = 0; kc < 8; ++kc) {
    int buf = kc & 1;
    if (kc + 1 < 8) stage(kc + 1, buf ^ 1);
    v16h bfrag;
    {
      const _Float16* br = &sB[buf][ml][0];
      int o2 = hi * 8;
      v4h* bp = (v4h*)&bfrag;
      bp[0] = *(const v4h*)(br + o2);
      bp[1] = *(const v4h*)(br + o2 + 4);
      bp[2] = *(const v4h*)(br + o2 + 16);
      bp[3] = *(const v4h*)(br + o2 + 20);
    }
    #pragma unroll
    for (int pass = 0; pass < 2; ++pass) {
      int tile = wave + pass * 8;
      const _Float16* ar = &sA[buf][tile][ml][0];
      int o1 = hi * 8;
      v16h afrag;
      v4h* ap = (v4h*)&afrag;
      ap[0] = *(const v4h*)(ar + o1);
      ap[1] = *(const v4h*)(ar + o1 + 4);
      ap[2] = *(const v4h*)(ar + o1 + 16);
      ap[3] = *(const v4h*)(ar + o1 + 20);
      acc[pass] = __builtin_amdgcn_wmma_f32_16x16x32_f16(false, afrag, false, bfrag,
                                                         (short)0, acc[pass], false, false);
    }
    wait_async_stage();
    __syncthreads();
  }

  // epilogue: bias + relu + logits weight, reduce over all 256 output channels
  int mrow0 = hi * 8;
  float part = 0.f;
  #pragma unroll
  for (int pass = 0; pass < 2; ++pass) {
    int coT = (wave + pass * 8) * 16;
    #pragma unroll
    for (int v = 0; v < 8; ++v) {
      int co = coT + mrow0 + v;
      float val = acc[pass][v] + dbias[co];
      val = val > 0.f ? val : 0.f;
      part += lw[co] * val;
    }
  }
  part += __shfl_xor(part, 16, 32);           // combine m rows 0-7 with 8-15
  if (lane < 16) sRed[wave][lane] = part;
  __syncthreads();
  if (tid < 16) {
    float tot = lb[0];
    #pragma unroll
    for (int w2 = 0; w2 < 8; ++w2) tot += sRed[w2][tid];
    int p = posbase + tid;
    int h = p / W_, w = p % W_;
    int kk = kl >> 1, ll = kl & 1;
    outp[(size_t)r * NN_ + (2 * h + kk) * 56 + (2 * w + ll)] = tot;
  }
}

// ---------------------------------------------------------------- launcher

extern "C" void kernel_launch(void* const* d_in, const int* in_sizes, int n_in,
                              void* d_out, int out_size, void* d_ws, size_t ws_size,
                              hipStream_t stream) {
  const float* feat_a   = (const float*)d_in[0];
  const float* mask_a   = (const float*)d_in[1];
  const float* feat_b   = (const float*)d_in[2];
  const float* mask_b   = (const float*)d_in[3];
  const float* pos_mu   = (const float*)d_in[4];
  const float* neg_mu   = (const float*)d_in[5];
  const float* conv1_w  = (const float*)d_in[6];
  const float* conv1_b  = (const float*)d_in[7];
  const float* conv2_w  = (const float*)d_in[8];
  const float* conv2_b  = (const float*)d_in[9];
  const float* convs_w  = (const float*)d_in[10];
  const float* convs_b  = (const float*)d_in[11];
  const float* deconv_w = (const float*)d_in[12];
  const float* deconv_b = (const float*)d_in[13];
  const float* logits_w = (const float*)d_in[14];
  const float* logits_b = (const float*)d_in[15];
  float* out = (float*)d_out;

  char* ws = (char*)d_ws;
  size_t off = 0;
  auto alloc = [&](size_t bytes) {
    size_t o = off;
    off += (bytes + 255) & ~(size_t)255;
    return o;
  };
  float*    mA     = (float*)(ws + alloc((size_t)R_ * N_ * 4));
  float*    mB     = (float*)(ws + alloc((size_t)R_ * N_ * 4));
  float*    gap    = (float*)(ws + alloc((size_t)R_ * C_ * 4));
  float*    co2    = (float*)(ws + alloc((size_t)R_ * C_ * 4));
  float*    mu     = (float*)(ws + alloc((size_t)R_ * C_ * 8 * 4));
  _Float16* featH  = (_Float16*)(ws + alloc((size_t)R_ * CIN1 * N_ * 2));
  _Float16* actA   = (_Float16*)(ws + alloc((size_t)R_ * C_ * N_ * 2));
  _Float16* actB   = (_Float16*)(ws + alloc((size_t)R_ * C_ * N_ * 2));
  _Float16* w1t    = (_Float16*)(ws + alloc((size_t)256 * 9 * KSTR1 * 2));
  _Float16* wst    = (_Float16*)(ws + alloc((size_t)3 * 256 * 9 * 256 * 2));
  _Float16* wdt    = (_Float16*)(ws + alloc((size_t)4 * 256 * 256 * 2));
  (void)ws_size; (void)in_sizes; (void)n_in; (void)out_size;

  auto gs = [](int total) { return (total + 255) / 256; };

  resize_masks_kernel<<<gs(2 * R_ * N_), 256, 0, stream>>>(mask_a, mask_b, mA, mB);
  cvt_w1_kernel<<<gs(256 * 9 * KSTR1), 256, 0, stream>>>(conv1_w, w1t);
  cvt_ws_kernel<<<gs(3 * 256 * 9 * 256), 256, 0, stream>>>(convs_w, wst);
  cvt_wd_kernel<<<gs(4 * 256 * 256), 256, 0, stream>>>(deconv_w, wdt);
  mu_init_kernel<<<gs(R_ * C_ * 8), 256, 0, stream>>>(pos_mu, neg_mu, mu);

  gap_kernel<<<R_, 256, 0, stream>>>(feat_a, mA, gap);
  conv2gap_kernel<<<R_, 256, 0, stream>>>(gap, conv2_w, conv2_b, conv1_b, co2);

  for (int s = 0; s < 6; ++s)
    em_stage_kernel<<<dim3(R_, 2), 256, 0, stream>>>(feat_a, mA, mu);   // pos/neg on frame a
  for (int s = 0; s < 6; ++s)
    em_stage_kernel<<<dim3(R_, 2), 256, 0, stream>>>(feat_b, mB, mu);   // refine on frame b

  copy_featb_kernel<<<gs(R_ * C_ * N_), 256, 0, stream>>>(feat_b, featH);
  prop_feat_kernel<<<R_, 256, 0, stream>>>(feat_b, mu, mA, featH);

  dim3 cgrid(49, R_);
  // conv1 (cin=259, K padded to 288) + per-(r,co) bias (conv1_b + conv2(gap) + conv2_b)
  conv3x3_wmma<<<cgrid, 256, 0, stream>>>(featH, w1t, CIN1, 9, KSTR1, co2, 1, 0, actA);
  // three 3x3 conv + relu layers
  conv3x3_wmma<<<cgrid, 256, 0, stream>>>(actA, wst + (size_t)0 * 256 * 9 * 256, C_, 8, 256,
                                          convs_b + 0, 0, 1, actB);
  conv3x3_wmma<<<cgrid, 256, 0, stream>>>(actB, wst + (size_t)1 * 256 * 9 * 256, C_, 8, 256,
                                          convs_b + 256, 0, 1, actA);
  conv3x3_wmma<<<cgrid, 256, 0, stream>>>(actA, wst + (size_t)2 * 256 * 9 * 256, C_, 8, 256,
                                          convs_b + 512, 0, 1, actB);
  // deconv 2x2 stride-2 + relu + logits 1x1, writes every output pixel (no atomics)
  deconv_logits_wmma<<<dim3(49, R_, 4), 256, 0, stream>>>(actB, wdt, deconv_b,
                                                          logits_w, logits_b, out);
}
